// DecoderSlayerSNN_36541581754609
// MI455X (gfx1250) — matrix operand order repaired and checked by reading, not compiled, not run on hardware
//
#include <hip/hip_runtime.h>

typedef __attribute__((ext_vector_type(16))) _Float16 v16h;
typedef __attribute__((ext_vector_type(8)))  float    v8f;
typedef __attribute__((ext_vector_type(4)))  float    f32x4;

#define B_    128
#define IN_   182
#define KPAD  192          // 6 * 32 (WMMA K-steps)
#define H_    256
#define O_    2
#define T_    2000
#define WPR   63           // u32 words per (b,h) spike-bit row: ceil(2000/32)

// ---------------------------------------------------------------------------
// K0: spike [B][IN][T] f32  ->  spikeT [T][B][KPAD] f16  (K zero-padded)
// ---------------------------------------------------------------------------
__global__ void k0_transpose(const float* __restrict__ spike,
                             _Float16* __restrict__ spikeT) {
  const int i  = threadIdx.x;           // 0..191
  const int t0 = blockIdx.x * 16;       // 125 chunks
  const int b  = blockIdx.y;            // 128
  float vals[16];
  if (i < IN_) {
    const f32x4* src4 = (const f32x4*)(spike + ((size_t)b * IN_ + i) * T_ + t0);
    #pragma unroll
    for (int q = 0; q < 4; ++q) {
      f32x4 x = src4[q];
      #pragma unroll
      for (int r = 0; r < 4; ++r) vals[q * 4 + r] = x[r];
    }
  } else {
    #pragma unroll
    for (int r = 0; r < 16; ++r) vals[r] = 0.0f;
  }
  #pragma unroll
  for (int r = 0; r < 16; ++r)
    spikeT[(size_t)(t0 + r) * (B_ * KPAD) + b * KPAD + i] = (_Float16)vals[r];
}

// ---------------------------------------------------------------------------
// K1: fused  z1 = W1 x spike  (12x v_wmma_f32_16x16x32_f16 per t, hi+lo split
// of W1 for ~f32 accuracy)  +  CUBA LIF scan  ->  packed spike bits.
// One wave per 16(h) x 16(b) tile. Only 128 waves exist (scan-limited), so
// there is at most one wave per SIMD: latency must be hidden *inside* the
// wave. Hence (a) __launch_bounds__(32,1) for the full VGPR file, and
// (b) manual 2-deep software pipeline: distinct ping-pong B-fragment register
// sets so the t+1 loads are in flight while the t WMMA chain executes
// (a single shared buffer made the compiler serialize load->wait0->wmma x6).
// ---------------------------------------------------------------------------
__global__ void __launch_bounds__(32, 1)
k1_layer1(const float* __restrict__ W1,
          const _Float16* __restrict__ spikeT,
          unsigned* __restrict__ bits) {
  const int lane = threadIdx.x;         // 32
  const int b0   = blockIdx.x * 16;     // 8 batch tiles
  const int h0   = blockIdx.y * 16;     // 16 hidden tiles
  const int col  = lane & 15;
  const int half = lane >> 4;           // 0 | 1
  const int hA   = h0 + col;            // A-matrix row owned by this lane

  // Preload W1 tile as hi/lo f16 WMMA A-fragments (exact to ~22 mantissa bits).
  v16h ah[6], al[6];
  #pragma unroll
  for (int kk = 0; kk < 6; ++kk) {
    #pragma unroll
    for (int e = 0; e < 16; ++e) {
      const int K = kk * 32 + (half ? 16 : 0) + e;
      float wf = (K < IN_) ? W1[hA * IN_ + K] : 0.0f;
      _Float16 whi = (_Float16)wf;
      _Float16 wlo = (_Float16)(wf - (float)whi);
      ah[kk][e] = whi;
      al[kk][e] = wlo;
    }
  }

  float    u[8], v[8];
  unsigned sb[8];
  #pragma unroll
  for (int j = 0; j < 8; ++j) { u[j] = 0.0f; v[j] = 0.0f; sb[j] = 0u; }

  // Per-lane base into spikeT row: contiguous 32B = one B-fragment slice.
  const _Float16* rowp = spikeT + (size_t)(b0 + col) * KPAD + (half ? 16 : 0);
  // Per-lane base into the packed spike-bit output; j*WPR becomes an
  // immediate store offset.
  unsigned* bitp = bits + ((size_t)(b0 + col) * H_ + h0 + 8 * half) * WPR;

  auto loadB = [&](v16h (&bf)[6], int t) {
    const _Float16* p = rowp + (size_t)t * (B_ * KPAD);
    #pragma unroll
    for (int kk = 0; kk < 6; ++kk) bf[kk] = *(const v16h*)(p + kk * 32);
  };

  auto step = [&](const v16h (&bf)[6], int t, bool mayFlush) {
    v8f c = {};
    #pragma unroll
    for (int kk = 0; kk < 6; ++kk) {
      c = __builtin_amdgcn_wmma_f32_16x16x32_f16(false, ah[kk], false, bf[kk],
                                                 (short)0, c, false, false);
      c = __builtin_amdgcn_wmma_f32_16x16x32_f16(false, al[kk], false, bf[kk],
                                                 (short)0, c, false, false);
    }
    // CUBA LIF: u = 0.95u + z ; v = 0.95v + u ; spike & hard reset.
    #pragma unroll
    for (int j = 0; j < 8; ++j) {
      u[j] = 0.95f * u[j] + c[j];
      v[j] = 0.95f * v[j] + u[j];
      const bool s = (v[j] >= 1.0f);
      v[j] = s ? 0.0f : v[j];
      sb[j] |= (s ? 1u : 0u) << (t & 31);
    }
    if (mayFlush && ((t & 31) == 31)) {   // uniform branch (t is scalar)
      const int w = t >> 5;
      #pragma unroll
      for (int j = 0; j < 8; ++j) {
        bitp[j * WPR + w] = sb[j];
        sb[j] = 0u;
      }
    }
  };

  v16h bf0[6], bf1[6];
  loadB(bf0, 0);
  for (int t = 0; t < T_; t += 2) {
    if (t + 8 < T_)   // -> global_prefetch_b8 of a future slab
      __builtin_prefetch(rowp + (size_t)(t + 8) * (B_ * KPAD), 0, 1);
    loadB(bf1, t + 1);            // in flight during step(t)
    step(bf0, t, false);          // even t never hits (t&31)==31
    if (t + 2 < T_) loadB(bf0, t + 2);  // in flight during step(t+1)
    step(bf1, t + 1, true);
  }
  // final partial word (t = 1984..1999 -> word 62)
  #pragma unroll
  for (int j = 0; j < 8; ++j) bitp[j * WPR + (WPR - 1)] = sb[j];
}

// ---------------------------------------------------------------------------
// K2: layer-2 synaptic input with axonal delay folded into the bit gather:
//     z2[b,o,t] = sum_h W2[o,h] * spikebit[b,h,t-d[h]]   (0 for t < d[h])
// W2 rows + delays staged in LDS. One wave = one (b, 32-t chunk).
// ---------------------------------------------------------------------------
__global__ void k2_layer2_syn(const unsigned* __restrict__ bits,
                              const float* __restrict__ W2,
                              const int* __restrict__ delay,
                              float* __restrict__ z2) {
  __shared__ float w0s[H_];
  __shared__ float w1s[H_];
  __shared__ int   dls[H_];
  const int lane = threadIdx.x;
  for (int h = lane; h < H_; h += 32) {
    w0s[h] = W2[h];
    w1s[h] = W2[H_ + h];
    dls[h] = delay[h];
  }
  __syncthreads();

  const int b = blockIdx.x;                 // 128
  const int t = blockIdx.y * 32 + lane;     // 63 chunks, t <= 2015 (guarded)
  const unsigned* rowb = bits + (size_t)b * H_ * WPR;

  float a0 = 0.0f, a1 = 0.0f;
  for (int h = 0; h < H_; ++h) {
    const int tt = t - dls[h];
    float sp = 0.0f;
    if (tt >= 0 && t < T_) {
      const unsigned wrd = rowb[h * WPR + (tt >> 5)];
      sp = (float)((wrd >> (tt & 31)) & 1u);
    }
    a0 += sp * w0s[h];
    a1 += sp * w1s[h];
  }
  if (t < T_) {
    z2[((size_t)b * T_ + t) * 2 + 0] = a0;
    z2[((size_t)b * T_ + t) * 2 + 1] = a1;
  }
}

// ---------------------------------------------------------------------------
// K3: layer-2 leaky integrator (no spike): u = 0.5u + z ; v = 0.05v + u.
// 256 independent scans (one lane each) -> out [B][O][T] f32.
// ---------------------------------------------------------------------------
__global__ void k3_scan2(const float* __restrict__ z2, float* __restrict__ out) {
  const int p = blockIdx.x * 32 + threadIdx.x;   // 0..255
  const int b = p >> 1, o = p & 1;
  const float* zp = z2 + (size_t)b * T_ * 2 + o;
  float* op = out + ((size_t)b * O_ + o) * T_;
  float u = 0.0f, v = 0.0f;
  for (int t = 0; t < T_; ++t) {
    const float z = zp[2 * t];
    u = 0.5f  * u + z;
    v = 0.05f * v + u;
    op[t] = v;
  }
}

// ---------------------------------------------------------------------------
extern "C" void kernel_launch(void* const* d_in, const int* in_sizes, int n_in,
                              void* d_out, int out_size, void* d_ws, size_t ws_size,
                              hipStream_t stream) {
  (void)in_sizes; (void)n_in; (void)out_size; (void)ws_size;
  const float* spike = (const float*)d_in[0];   // [128,182,2000] f32
  const float* W1    = (const float*)d_in[1];   // [256,182]      f32
  const float* W2    = (const float*)d_in[2];   // [2,256]        f32
  const int*   delay = (const int*)  d_in[3];   // [256]          i32
  float* out = (float*)d_out;                   // [128,2,2000]   f32

  char* ws = (char*)d_ws;
  _Float16* spikeT = (_Float16*)ws;                         // 98,304,000 B
  const size_t spikeT_bytes = (size_t)T_ * B_ * KPAD * sizeof(_Float16);
  unsigned* bits = (unsigned*)(ws + spikeT_bytes);          //  8,257,536 B
  const size_t bits_bytes = (size_t)B_ * H_ * WPR * sizeof(unsigned);
  float* z2 = (float*)(ws + spikeT_bytes + bits_bytes);     //  2,048,000 B

  k0_transpose <<<dim3(T_ / 16, B_), 192, 0, stream>>>(spike, spikeT);
  k1_layer1    <<<dim3(B_ / 16, H_ / 16), 32, 0, stream>>>(W1, spikeT, bits);
  k2_layer2_syn<<<dim3(B_, WPR), 32, 0, stream>>>(bits, W2, delay, z2);
  k3_scan2     <<<dim3(8), 32, 0, stream>>>(z2, out);
}
